// GCNClassifier_58780922413863
// MI455X (gfx1250) — matrix operand order repaired and checked by reading, not compiled, not run on hardware
//
#include <hip/hip_runtime.h>

// ---------------------------------------------------------------------------
// GCN 2-layer forward for MI455X (gfx1250, wave32).
//  - GEMMs via v_wmma_f32_16x16x32_f16 (f16 in, f32 accum); weights are
//    pre-transposed so BOTH A and B fragments are contiguous b128 loads, and
//    all NT B-tiles of a k-step are loaded before the WMMA burst so the
//    loads overlap (one s_wait per k-step instead of per WMMA).
//  - edge propagation via f32 global atomics (whole working set < 192 MB L2),
//    one wave per edge, 128-bit per-lane feature accesses.
// ---------------------------------------------------------------------------

typedef __attribute__((ext_vector_type(16))) _Float16 v16h;
typedef __attribute__((ext_vector_type(8)))  _Float16 v8h;
typedef __attribute__((ext_vector_type(8)))  float    v8f;

// ----------------------------- elementwise ---------------------------------

__global__ __launch_bounds__(256) void k_fill_f32(float* __restrict__ dst,
                                                  float v, int n) {
    int i = blockIdx.x * blockDim.x + threadIdx.x;
    if (i < n) dst[i] = v;
}

__global__ __launch_bounds__(256) void k_count_deg(const int* __restrict__ col,
                                                   float* __restrict__ deg, int E) {
    int e = blockIdx.x * blockDim.x + threadIdx.x;
    if (e < E) unsafeAtomicAdd(&deg[col[e]], 1.0f);
}

__global__ __launch_bounds__(256) void k_rsqrt_inplace(float* __restrict__ d, int n) {
    int i = blockIdx.x * blockDim.x + threadIdx.x;
    if (i < n) d[i] = rsqrtf(d[i]);   // deg >= 1 always (self loop)
}

__global__ __launch_bounds__(256) void k_cvt_f32_f16(const float* __restrict__ src,
                                                     _Float16* __restrict__ dst, int n) {
    int i = blockIdx.x * blockDim.x + threadIdx.x;
    if (i < n) dst[i] = (_Float16)src[i];
}

// dst[Nc x K] = transpose(src[K x Nc]) downconverted to f16 (weights, <=64KB)
__global__ __launch_bounds__(256) void k_transpose_cvt(const float* __restrict__ src,
                                                       _Float16* __restrict__ dst,
                                                       int K, int Nc) {
    int i = blockIdx.x * blockDim.x + threadIdx.x;
    if (i < K * Nc) {
        int k = i / Nc, n = i - k * Nc;
        dst[(size_t)n * K + k] = (_Float16)src[i];
    }
}

// out[i] = bias[i & mask]   (broadcast bias into the scatter target)
__global__ __launch_bounds__(256) void k_bias_bcast(float* __restrict__ dst,
                                                    const float* __restrict__ bias,
                                                    int mask, int n) {
    int i = blockIdx.x * blockDim.x + threadIdx.x;
    if (i < n) dst[i] = bias[i & mask];
}

// h1h = f16(relu(agg + b1))  (fused bias + relu + downconvert)
__global__ __launch_bounds__(256) void k_bias_relu_cvt(const float* __restrict__ agg,
                                                       const float* __restrict__ bias,
                                                       _Float16* __restrict__ dst,
                                                       int mask, int n) {
    int i = blockIdx.x * blockDim.x + threadIdx.x;
    if (i < n) {
        float v = agg[i] + bias[i & mask];
        dst[i] = (_Float16)fmaxf(v, 0.0f);
    }
}

// ------------------------------- WMMA GEMM ---------------------------------
// C[M x Nc] = A[M x K] * Bt^T, with Bt = B transposed, i.e. Bt[Nc x K].
// One wave computes a 16 x Nc stripe (NT = Nc/16 column tiles), so the A
// fragment is loaded once per k-step and reused across NT WMMAs.
// A fragment (16-bit A 16x32 layout): lane l holds row l&15, K-chunks
// [kb,kb+8) and [kb+16,kb+24), kb=(l>>4)*8  -> two contiguous b128 loads.
// B fragment (16-bit B 32x16 layout): lane l holds column l&15, K-chunks
// [kbB,kbB+8) and [kbB+16,kbB+24), kbB=(l>>4)*8 -> two b128 loads of Bt row.
// Requires M%16==0, Nc%16==0, K%32==0.
template <int K, int NT>
__global__ __launch_bounds__(32)
void k_gemm_wmma_f16(const _Float16* __restrict__ A,
                     const _Float16* __restrict__ Bt,
                     float* __restrict__ C) {
    constexpr int Nc = NT * 16;
    const int m0   = blockIdx.x << 4;
    const int lane = threadIdx.x;      // 0..31, EXEC all-ones (no divergence)
    const int hs   = lane >> 4;        // lane-half select
    const int l    = lane & 15;

    v8f acc[NT];
    #pragma unroll
    for (int t = 0; t < NT; ++t) acc[t] = v8f{};

    const int kbA = hs * 8;
    const int kbB = hs * 8;

    const _Float16* arow = A  + (size_t)(m0 + l) * K;
    const _Float16* brow = Bt + (size_t)l * K + kbB;

    #pragma unroll
    for (int k0 = 0; k0 < K; k0 += 32) {
        // A fragment: two contiguous 16-byte chunks of row (m0+l)
        v8h alo = *(const v8h*)(arow + k0 + kbA);
        v8h ahi = *(const v8h*)(arow + k0 + kbA + 16);
        v16h a;
        #pragma unroll
        for (int i = 0; i < 8; ++i) { a[i] = alo[i]; a[i + 8] = ahi[i]; }

        // Load ALL NT B fragments first (distinct regs -> overlapping loads)
        v16h b[NT];
        #pragma unroll
        for (int t = 0; t < NT; ++t) {
            const _Float16* bp = brow + (size_t)(t * 16) * K + k0;
            v8h blo = *(const v8h*)(bp);
            v8h bhi = *(const v8h*)(bp + 16);
            #pragma unroll
            for (int i = 0; i < 8; ++i) { b[t][i] = blo[i]; b[t][i + 8] = bhi[i]; }
        }

        // WMMA burst over the column tiles
        #pragma unroll
        for (int t = 0; t < NT; ++t) {
            acc[t] = __builtin_amdgcn_wmma_f32_16x16x32_f16(
                /*neg_a=*/false, a, /*neg_b=*/false, b[t],
                /*c_mod=*/(short)0, acc[t], /*reuse_a=*/false, /*reuse_b=*/false);
        }
    }

    // C/D layout: VGPR v -> row v + 8*hs, col = l (within each 16-col tile)
    #pragma unroll
    for (int t = 0; t < NT; ++t) {
        float* crow = C + (size_t)(m0 + 8 * hs) * Nc + t * 16 + l;
        #pragma unroll
        for (int v = 0; v < 8; ++v) crow[(size_t)v * Nc] = acc[t][v];
    }
}

// ----------------------------- edge scatter --------------------------------
// One wave (32 lanes) per edge; lane owns F/32 consecutive features.
// Edges [0,E) come from row/col arrays; edges [E, E+Nn) are self loops.
template <int F>
__global__ __launch_bounds__(256)
void k_scatter_add(const float* __restrict__ h,
                   const float* __restrict__ dinv,
                   const int* __restrict__ row,
                   const int* __restrict__ col,
                   int E, int Nn,
                   float* __restrict__ out) {
    constexpr int VEC = F / 32;
    long long tid = (long long)blockIdx.x * blockDim.x + threadIdx.x;
    int e    = (int)(tid >> 5);
    int lane = (int)(tid & 31);
    if (e >= E + Nn) return;

    int r, c;
    if (e < E) { r = row[e]; c = col[e]; }
    else       { r = e - E;  c = r;      }

    float norm = dinv[r] * dinv[c];

    const float* src = h   + (size_t)r * F + lane * VEC;
    float*       dst = out + (size_t)c * F + lane * VEC;
    #pragma unroll
    for (int j = 0; j < VEC; ++j)
        unsafeAtomicAdd(dst + j, src[j] * norm);   // global_atomic_add_f32
}

// ------------------------------- launcher ----------------------------------

extern "C" void kernel_launch(void* const* d_in, const int* in_sizes, int n_in,
                              void* d_out, int out_size, void* d_ws, size_t ws_size,
                              hipStream_t stream) {
    const float* x  = (const float*)d_in[0];
    const int*   ei = (const int*)  d_in[1];   // [2, E] flat: row = ei, col = ei+E
    const float* W1 = (const float*)d_in[2];
    const float* b1 = (const float*)d_in[3];
    const float* W2 = (const float*)d_in[4];
    const float* b2 = (const float*)d_in[5];

    const int E    = in_sizes[1] / 2;          // 800000
    const int Fh   = in_sizes[3];              // 128
    const int Fout = in_sizes[5];              // 64
    const int Fin  = in_sizes[2] / Fh;         // 128
    const int N    = in_sizes[0] / Fin;        // 50000

    // ---- workspace carve-out (~65 MB; buffers reused across phases) ----
    char* p = (char*)d_ws;
    auto carve = [&](size_t bytes) {
        char* r = p; p += (bytes + 255) & ~(size_t)255; return r;
    };
    float*    deg  = (float*)   carve((size_t)N * 4);            // -> dinv in place
    _Float16* xh   = (_Float16*)carve((size_t)N * Fin * 2);      // reused as h1h
    _Float16* w1t  = (_Float16*)carve((size_t)Fin * Fh * 2);     // W1^T  [Fh x Fin]
    _Float16* w2t  = (_Float16*)carve((size_t)Fh * Fout * 2);    // W2^T  [Fout x Fh]
    float*    h0   = (float*)   carve((size_t)N * Fh * 4);       // reused as h2
    float*    agg1 = (float*)   carve((size_t)N * Fh * 4);

    const int*   row = ei;
    const int*   col = ei + E;
    float*       out = (float*)d_out;

    const int nxh  = N * Fin;      // 6.4M
    const int nagg = N * Fh;       // 6.4M
    const int nout = N * Fout;     // 3.2M
    auto blks = [](long long n, int b) { return (unsigned)((n + b - 1) / b); };

    // 1) degree (self loop contributes 1), then dinv = rsqrt(deg) in place
    k_fill_f32<<<blks(N, 256), 256, 0, stream>>>(deg, 1.0f, N);
    k_count_deg<<<blks(E, 256), 256, 0, stream>>>(col, deg, E);
    k_rsqrt_inplace<<<blks(N, 256), 256, 0, stream>>>(deg, N);

    // 2) downconvert x; transpose+downconvert W1, W2 (contiguous B fragments)
    k_cvt_f32_f16<<<blks(nxh, 256), 256, 0, stream>>>(x, xh, nxh);
    k_transpose_cvt<<<blks(Fin * Fh, 256), 256, 0, stream>>>(W1, w1t, Fin, Fh);
    k_transpose_cvt<<<blks(Fh * Fout, 256), 256, 0, stream>>>(W2, w2t, Fh, Fout);

    // 3) h0 = x @ W1  (each wave: 16 x 128 stripe, 32 unrolled v_wmma)
    k_gemm_wmma_f16<128, 8><<<N / 16, 32, 0, stream>>>(xh, w1t, h0);

    // 4) layer-1 propagation: agg1 = scatter_add(norm * h0[row] -> col)
    k_fill_f32<<<blks(nagg, 256), 256, 0, stream>>>(agg1, 0.0f, nagg);
    k_scatter_add<128><<<blks((long long)(E + N) * 32, 256), 256, 0, stream>>>(
        h0, deg, row, col, E, N, agg1);

    // 5) h1 = f16(relu(agg1 + b1))   (written into xh, which is now free)
    k_bias_relu_cvt<<<blks(nagg, 256), 256, 0, stream>>>(agg1, b1, xh, Fh - 1, nagg);

    // 6) h2 = h1 @ W2  (each wave: 16 x 64 stripe; into h0's storage as [N,64])
    k_gemm_wmma_f16<128, 4><<<N / 16, 32, 0, stream>>>(xh, w2t, h0);

    // 7) layer-2 propagation directly into d_out, bias pre-broadcast
    k_bias_bcast<<<blks(nout, 256), 256, 0, stream>>>(out, b2, Fout - 1, nout);
    k_scatter_add<64><<<blks((long long)(E + N) * 32, 256), 256, 0, stream>>>(
        h0, deg, row, col, E, N, out);
}
